// AttentionBlock_20916490731606
// MI455X (gfx1250) — matrix operand order repaired and checked by reading, not compiled, not run on hardware
//
#include <hip/hip_runtime.h>
#include <hip/hip_bf16.h>

// ---------------------------------------------------------------------------
// Types for CDNA5 WMMA
// ---------------------------------------------------------------------------
typedef __attribute__((ext_vector_type(8)))  float  v8f;
typedef __attribute__((ext_vector_type(16))) __bf16 v16bf;
typedef __attribute__((ext_vector_type(8)))  __bf16 v8bf;
typedef __attribute__((__vector_size__(4 * sizeof(int)))) int v4i;

__device__ __forceinline__ v16bf cat16(v8bf a, v8bf b) {
    return __builtin_shufflevector(a, b, 0,1,2,3,4,5,6,7,8,9,10,11,12,13,14,15);
}

__device__ __forceinline__ v8f wmma_bf16(v16bf a, v16bf b, v8f c) {
    // D = A(16x32 bf16) x B(32x16 bf16) + C(16x16 f32)
    return __builtin_amdgcn_wmma_f32_16x16x32_bf16(false, a, false, b,
                                                   (short)0, c, false, false);
}

__device__ __forceinline__ float lane_bcast(float v, int srcLane) {
    return __int_as_float(__builtin_amdgcn_ds_bpermute(srcLane << 2,
                                                       __float_as_int(v)));
}

// Async global->LDS copy (ASYNCcnt-tracked on gfx1250); sync fallback.
__device__ __forceinline__ void async_copy_b128(const __bf16* g, __bf16* l) {
#if __has_builtin(__builtin_amdgcn_global_load_async_to_lds_b128)
    __builtin_amdgcn_global_load_async_to_lds_b128(
        (__attribute__((address_space(1))) v4i*)g,
        (__attribute__((address_space(3))) v4i*)l, 0, 0);
#else
    *(v8bf*)l = *(const v8bf*)g;
#endif
}

__device__ __forceinline__ void wait_async0() {
#if __has_builtin(__builtin_amdgcn_global_load_async_to_lds_b128)
#if __has_builtin(__builtin_amdgcn_s_wait_asynccnt)
    __builtin_amdgcn_s_wait_asynccnt(0);
#else
    asm volatile("s_wait_asynccnt 0x0" ::: "memory");
#endif
#endif
}

// Problem constants
#define BB     4
#define NN     4096        // H*W
#define CC     256
#define HEADS  4
#define HD     64
#define BN     (BB * NN)   // 16384 rows
#define GROUPS 8
#define CPG    32
#define RED_N  (NN * CPG)  // 131072 elements per (b,g)
#define LOG2E  1.44269504088896340736f

// ---------------------------------------------------------------------------
// Kernel 1: GroupNorm statistics (mean, rstd) per (b, group)
// ---------------------------------------------------------------------------
__global__ void gn_stats_kernel(const float* __restrict__ x,
                                float* __restrict__ stats) {
    __shared__ float r1[256];
    __shared__ float r2[256];
    int bg = blockIdx.x;            // 0..31
    int b = bg >> 3, g = bg & 7;
    const float* base = x + (size_t)b * NN * CC + g * CPG;
    int t = threadIdx.x;
    int c = t & 31;                 // channel within group
    float s = 0.f, s2 = 0.f;
    for (int p = (t >> 5); p < NN; p += 8) {
        float v = base[(size_t)p * CC + c];
        s += v; s2 += v * v;
    }
    r1[t] = s; r2[t] = s2;
    __syncthreads();
    for (int off = 128; off > 0; off >>= 1) {
        if (t < off) { r1[t] += r1[t + off]; r2[t] += r2[t + off]; }
        __syncthreads();
    }
    if (t == 0) {
        float inv = 1.0f / (float)RED_N;
        float mean = r1[0] * inv;
        float var = r2[0] * inv - mean * mean;
        stats[2 * bg]     = mean;
        stats[2 * bg + 1] = rsqrtf(var + 1e-5f);
    }
}

// ---------------------------------------------------------------------------
// Kernel 2: apply GN -> xn f32 (residual) + bf16 (GEMM input)
// ---------------------------------------------------------------------------
__global__ void gn_apply_kernel(const float* __restrict__ x,
                                const float* __restrict__ gscale,
                                const float* __restrict__ gbias,
                                const float* __restrict__ stats,
                                float* __restrict__ xnf,
                                __bf16* __restrict__ xnb) {
    int base = (blockIdx.x * 256 + threadIdx.x) * 8;
#pragma unroll
    for (int j = 0; j < 8; ++j) {
        int gi = base + j;
        int c = gi & 255;
        int sidx = ((gi >> 20) << 3) | ((gi >> 5) & 7);  // b*8 + group
        float mean = stats[2 * sidx];
        float rstd = stats[2 * sidx + 1];
        float v = (x[gi] - mean) * rstd * gscale[c] + gbias[c];
        xnf[gi] = v;
        xnb[gi] = (__bf16)v;
    }
}

// ---------------------------------------------------------------------------
// Kernel 3: convert weights to bf16
// ---------------------------------------------------------------------------
__global__ void wconv_kernel(const float* __restrict__ w_qkv,
                             const float* __restrict__ w_out,
                             __bf16* __restrict__ wqb,
                             __bf16* __restrict__ wob) {
    int i = blockIdx.x * 256 + threadIdx.x;   // 0 .. 262143
    if (i < CC * 3 * CC) {
        wqb[i] = (__bf16)w_qkv[i];
    } else {
        int j = i - CC * 3 * CC;
        wob[j] = (__bf16)w_out[j];
    }
}

// ---------------------------------------------------------------------------
// Kernel 4/6: WMMA GEMM  [BN x 256] x [256 x Ncols], 128x64 block tile.
// mode 0: QKV epilogue (scatter q/k/v bf16, q pre-scaled by log2e/16)
// mode 1: out-proj epilogue (+bias +residual, f32 out)
// ---------------------------------------------------------------------------
__global__ void gemm_bf16_kernel(const __bf16* __restrict__ A,
                                 const __bf16* __restrict__ W,
                                 int Ncols, int mode,
                                 __bf16* __restrict__ qb,
                                 __bf16* __restrict__ kb,
                                 __bf16* __restrict__ vb,
                                 const float* __restrict__ bias,
                                 const float* __restrict__ resid,
                                 float* __restrict__ out) {
    __shared__ __align__(16) __bf16 As[128 * 32];  // [row][k]  8 KB
    __shared__ __align__(16) __bf16 Bs[64 * 32];   // [col][k]  4 KB
    int t = threadIdx.x, lane = t & 31, w = t >> 5;
    int hi = lane >> 4, lc = lane & 15;
    int rowbase = blockIdx.x * 128, colbase = blockIdx.y * 64;

    v8f acc[4];
#pragma unroll
    for (int nc = 0; nc < 4; ++nc) acc[nc] = (v8f)0.0f;

    for (int kc = 0; kc < 8; ++kc) {
        __syncthreads();
        {   // stage A: 128 rows x 32 k (bf16), 32 B per thread (async)
            int row = t >> 1, half = t & 1;
            const __bf16* src = A + (size_t)(rowbase + row) * CC
                                  + kc * 32 + half * 16;
            __bf16* dst = As + row * 32 + half * 16;
            async_copy_b128(src, dst);
            async_copy_b128(src + 8, dst + 8);
        }
        {   // stage B transposed: Bs[col][k]
            int col = t >> 2, ks = (t & 3) * 8;
#pragma unroll
            for (int j = 0; j < 8; ++j)
                Bs[col * 32 + ks + j] =
                    W[(size_t)(kc * 32 + ks + j) * Ncols + colbase + col];
        }
        wait_async0();
        __syncthreads();

        // A operand: row = w*16 + lc
        const __bf16* ar = As + (w * 16 + lc) * 32;
        v16bf Aop = cat16(*(const v8bf*)(ar + hi * 8),
                          *(const v8bf*)(ar + 16 + hi * 8));
#pragma unroll
        for (int nc = 0; nc < 4; ++nc) {
            const __bf16* br = Bs + (nc * 16 + lc) * 32;
            v16bf Bop = cat16(*(const v8bf*)(br + hi * 8),
                              *(const v8bf*)(br + 16 + hi * 8));
            acc[nc] = wmma_bf16(Aop, Bop, acc[nc]);
        }
    }

    // epilogue
#pragma unroll
    for (int nc = 0; nc < 4; ++nc) {
#pragma unroll
        for (int r = 0; r < 8; ++r) {
            int grow = rowbase + w * 16 + r + 8 * hi;
            int gcol = colbase + nc * 16 + lc;
            float val = acc[nc][r];
            if (mode == 0) {
                int part = gcol >> 8;            // 0=q 1=k 2=v
                int within = gcol & 255;
                int head = within >> 6, dim = within & 63;
                int bb = grow >> 12, n = grow & 4095;
                size_t off = ((size_t)(bb * HEADS + head) * NN + n) * HD + dim;
                __bf16* dst = (part == 0) ? qb : (part == 1) ? kb : vb;
                // q carries softmax scale C^-0.5 = 1/16 and log2(e) for
                // exp2-domain online softmax
                float sv = (part == 0) ? val * (0.0625f * LOG2E) : val;
                dst[off] = (__bf16)sv;
            } else {
                size_t off = (size_t)grow * CC + gcol;
                out[off] = val + bias[gcol] + resid[off];
            }
        }
    }
}

// ---------------------------------------------------------------------------
// Kernel 5: flash attention.  grid = (4096/128, B*HEADS), 256 threads.
// Each wave: 16 query rows; loop over 4096 keys in chunks of 32.
// S^T tiles via WMMA (A=K rows, B=Q^T) so a query's scores live in 2 lanes;
// exp2(S^T) values land directly in the A-operand layout for the P*V WMMA.
// K staged with async global->LDS loads, double-buffered so chunk i+1
// streams in while chunk i is computed; V's transpose-staging global loads
// issue before compute, ds_stores after.
// ---------------------------------------------------------------------------
__global__ void flash_attn_kernel(const __bf16* __restrict__ Q,
                                  const __bf16* __restrict__ K,
                                  const __bf16* __restrict__ V,
                                  __bf16* __restrict__ res) {
    __shared__ __align__(16) __bf16 klds[2][32 * 64];  // [buf][key][dim] 8 KB
    __shared__ __align__(16) __bf16 vt[2][64 * 32];    // [buf][dim][key] 8 KB
    int t = threadIdx.x, lane = t & 31, w = t >> 5;
    int hi = lane >> 4, lc = lane & 15;
    int bh = blockIdx.y;
    int qbase = blockIdx.x * 128 + w * 16;

    const __bf16* qp = Q + (size_t)bh * NN * HD;
    const __bf16* kp = K + (size_t)bh * NN * HD;
    const __bf16* vp = V + (size_t)bh * NN * HD;

    // Q as B-operand (col = query lc, K-dim = head dims), held in VGPRs
    const __bf16* qr = qp + (size_t)(qbase + lc) * HD;
    v16bf Qb0 = cat16(*(const v8bf*)(qr + hi * 8),
                      *(const v8bf*)(qr + 16 + hi * 8));        // dims 0-31
    v16bf Qb1 = cat16(*(const v8bf*)(qr + 32 + hi * 8),
                      *(const v8bf*)(qr + 48 + hi * 8));   // dims 32-63

    v8f o[4];
#pragma unroll
    for (int nc = 0; nc < 4; ++nc) o[nc] = (v8f)0.0f;
    float m = -1.0e30f, lsum = 0.0f;

    const int vrow = t >> 3;          // key row handled by this thread
    const int vc0 = (t & 7) * 8;      // dim start

    // ---- prologue: stage chunk 0 into buffer 0 ----
    async_copy_b128(kp + t * 8, &klds[0][t * 8]);
    {
        v8bf vv = *(const v8bf*)(vp + (size_t)vrow * HD + vc0);
#pragma unroll
        for (int j = 0; j < 8; ++j)
            vt[0][(vc0 + j) * 32 + vrow] = vv[j];
    }
    wait_async0();

    for (int kc = 0; kc < NN / 32; ++kc) {
        int cur = kc & 1;
        __syncthreads();   // chunk kc staged; buffer cur^1 free for reuse

        // ---- prefetch chunk kc+1 into the other buffer ----
        bool more = (kc + 1 < NN / 32);
        v8bf vv;
        if (more) {
            size_t nb = (size_t)(kc + 1) * 32 * HD;
            async_copy_b128(kp + nb + t * 8, &klds[cur ^ 1][t * 8]);
            vv = *(const v8bf*)(vp + nb + (size_t)vrow * HD + vc0);
        }

        // ---- S^T tiles: [16 keys x 16 queries], two tiles cover 32 keys ----
        const __bf16* kbuf = klds[cur];
        v8f st0, st1;
        {
            const __bf16* kr = kbuf + lc * 64;            // keys 0-15
            v16bf Ka0 = cat16(*(const v8bf*)(kr + hi * 8),
                              *(const v8bf*)(kr + 16 + hi * 8));
            v16bf Ka1 = cat16(*(const v8bf*)(kr + 32 + hi * 8),
                              *(const v8bf*)(kr + 48 + hi * 8));
            v8f z = (v8f)0.0f;
            st0 = wmma_bf16(Ka0, Qb0, z);
            st0 = wmma_bf16(Ka1, Qb1, st0);
        }
        {
            const __bf16* kr = kbuf + (16 + lc) * 64;     // keys 16-31
            v16bf Ka0 = cat16(*(const v8bf*)(kr + hi * 8),
                              *(const v8bf*)(kr + 16 + hi * 8));
            v16bf Ka1 = cat16(*(const v8bf*)(kr + 32 + hi * 8),
                              *(const v8bf*)(kr + 48 + hi * 8));
            v8f z = (v8f)0.0f;
            st1 = wmma_bf16(Ka0, Qb0, z);
            st1 = wmma_bf16(Ka1, Qb1, st1);
        }

        // ---- online softmax in exp2 domain (scale folded into q) ----
        float cm = st0[0];
#pragma unroll
        for (int i = 1; i < 8; ++i) cm = fmaxf(cm, st0[i]);
#pragma unroll
        for (int i = 0; i < 8; ++i) cm = fmaxf(cm, st1[i]);
        cm = fmaxf(cm, lane_bcast(cm, lane ^ 16));
        float mnew = fmaxf(m, cm);
        float alpha = exp2f(m - mnew);
        float p0[8], p1[8], cs = 0.0f;
#pragma unroll
        for (int i = 0; i < 8; ++i) { p0[i] = exp2f(st0[i] - mnew); cs += p0[i]; }
#pragma unroll
        for (int i = 0; i < 8; ++i) { p1[i] = exp2f(st1[i] - mnew); cs += p1[i]; }
        cs += lane_bcast(cs, lane ^ 16);
        lsum = lsum * alpha + cs;
        m = mnew;

        // rescale accumulator: row r -> query (r + 8*hi), broadcast alpha
        float ab[8];
#pragma unroll
        for (int r = 0; r < 8; ++r) ab[r] = lane_bcast(alpha, r + 8 * hi);
#pragma unroll
        for (int nc = 0; nc < 4; ++nc)
#pragma unroll
            for (int r = 0; r < 8; ++r) o[nc][r] *= ab[r];

        // P already in A-operand layout: elems 0-7 = tile0, 8-15 = tile1
        v16bf Pa;
#pragma unroll
        for (int i = 0; i < 8; ++i) {
            Pa[i]     = (__bf16)p0[i];
            Pa[8 + i] = (__bf16)p1[i];
        }

        // ---- O += P * V  (B-operand from transposed V in LDS) ----
        const __bf16* vbuf = vt[cur];
#pragma unroll
        for (int nc = 0; nc < 4; ++nc) {
            const __bf16* vr = vbuf + (nc * 16 + lc) * 32;
            v16bf Vb = cat16(*(const v8bf*)(vr + hi * 8),
                             *(const v8bf*)(vr + 16 + hi * 8));
            o[nc] = wmma_bf16(Pa, Vb, o[nc]);
        }

        // ---- finish staging chunk kc+1 (V transpose stores) ----
        if (more) {
#pragma unroll
            for (int j = 0; j < 8; ++j)
                vt[cur ^ 1][(vc0 + j) * 32 + vrow] = vv[j];
        }
        wait_async0();
    }

    // finalize: divide by row sum, scatter to res[b, n, h*64 + dim] (bf16)
    float rl = 1.0f / lsum;
    float rb[8];
#pragma unroll
    for (int r = 0; r < 8; ++r) rb[r] = lane_bcast(rl, r + 8 * hi);
    int b = bh >> 2, h = bh & 3;
#pragma unroll
    for (int nc = 0; nc < 4; ++nc) {
#pragma unroll
        for (int r = 0; r < 8; ++r) {
            int n = qbase + r + 8 * hi;
            size_t off = ((size_t)(b * NN + n)) * CC + h * HD + nc * 16 + lc;
            res[off] = (__bf16)(o[nc][r] * rb[r]);
        }
    }
}

// ---------------------------------------------------------------------------
// Host launcher
// ---------------------------------------------------------------------------
extern "C" void kernel_launch(void* const* d_in, const int* in_sizes, int n_in,
                              void* d_out, int out_size, void* d_ws, size_t ws_size,
                              hipStream_t stream) {
    const float* x      = (const float*)d_in[0];
    const float* gscale = (const float*)d_in[1];
    const float* gbias  = (const float*)d_in[2];
    const float* w_qkv  = (const float*)d_in[3];
    const float* w_out  = (const float*)d_in[4];
    const float* b_out  = (const float*)d_in[5];
    float* out = (float*)d_out;

    char* ws = (char*)d_ws;
    float*  stats = (float*)(ws + 0);                 //    256 B
    float*  xnf   = (float*)(ws + 256);               // 16 MB f32 xn
    __bf16* xnb   = (__bf16*)(ws + 16777472);         //  8 MB bf16 xn
    __bf16* wqb   = (__bf16*)(ws + 25166080);         // 384 KB
    __bf16* wob   = (__bf16*)(ws + 25559296);         // 128 KB
    __bf16* qb    = (__bf16*)(ws + 25690368);         //  8 MB
    __bf16* kb    = (__bf16*)(ws + 34078976);         //  8 MB
    __bf16* vb    = (__bf16*)(ws + 42467584);         //  8 MB
    __bf16* resb  = (__bf16*)(ws + 50856192);         //  8 MB  (total ~56.5 MB)

    // 1. GroupNorm statistics
    gn_stats_kernel<<<dim3(BB * GROUPS), dim3(256), 0, stream>>>(x, stats);
    // 2. Apply GN
    gn_apply_kernel<<<dim3((BN * CC) / (256 * 8)), dim3(256), 0, stream>>>(
        x, gscale, gbias, stats, xnf, xnb);
    // 3. Weights -> bf16
    wconv_kernel<<<dim3((CC * 3 * CC + CC * CC) / 256), dim3(256), 0, stream>>>(
        w_qkv, w_out, wqb, wob);
    // 4. QKV projection GEMM (q pre-scaled in epilogue)
    gemm_bf16_kernel<<<dim3(BN / 128, (3 * CC) / 64), dim3(256), 0, stream>>>(
        xnb, wqb, 3 * CC, 0, qb, kb, vb, nullptr, nullptr, nullptr);
    // 5. Flash attention
    flash_attn_kernel<<<dim3(NN / 128, BB * HEADS), dim3(256), 0, stream>>>(
        qb, kb, vb, resb);
    // 6. Output projection + bias + residual
    gemm_bf16_kernel<<<dim3(BN / 128, CC / 64), dim3(256), 0, stream>>>(
        resb, wob, CC, 1, nullptr, nullptr, nullptr, b_out, xnf, out);
}
